// MotionMamba_52836687675944
// MI455X (gfx1250) — matrix operand order repaired and checked by reading, not compiled, not run on hardware
//
#include <hip/hip_runtime.h>
#include <hip/hip_bf16.h>
#include <math.h>

#define D_MODEL 512
#define D_INNER 1024
#define D_STATE 16
#define D_CONV  4
#define DT_RANK 32
#define DEPTH   24
#define BATCH   4
#define SEQLEN  196
#define ROWS    (BATCH * SEQLEN)   // 784 = 49 * 16
#define LN_EPS  1e-5f

typedef __attribute__((ext_vector_type(16))) __bf16 v16bf;
typedef __attribute__((ext_vector_type(8)))  float  v8f;

// ---------------------------------------------------------------------------
// residual add + layernorm (one block per row, 512 cols, 256 threads x2)
// ---------------------------------------------------------------------------
__global__ void add_ln_kernel(const float* __restrict__ xin,
                              const float* __restrict__ hidden,
                              float* __restrict__ residual,
                              float* __restrict__ out,
                              const float* __restrict__ w,
                              const float* __restrict__ b,
                              int first) {
  __shared__ float s1[256], s2[256];
  const int row = blockIdx.x;
  const int tid = threadIdx.x;
  float v0, v1;
  if (first) {
    v0 = xin[row * D_MODEL + tid];
    v1 = xin[row * D_MODEL + tid + 256];
  } else {
    v0 = residual[row * D_MODEL + tid]       + hidden[row * D_MODEL + tid];
    v1 = residual[row * D_MODEL + tid + 256] + hidden[row * D_MODEL + tid + 256];
  }
  residual[row * D_MODEL + tid]       = v0;
  residual[row * D_MODEL + tid + 256] = v1;
  s1[tid] = v0 + v1;
  s2[tid] = v0 * v0 + v1 * v1;
  __syncthreads();
  for (int s = 128; s > 0; s >>= 1) {
    if (tid < s) { s1[tid] += s1[tid + s]; s2[tid] += s2[tid + s]; }
    __syncthreads();
  }
  const float mean = s1[0] * (1.0f / D_MODEL);
  const float var  = s2[0] * (1.0f / D_MODEL) - mean * mean;
  const float rstd = rsqrtf(var + LN_EPS);
  out[row * D_MODEL + tid]       = (v0 - mean) * rstd * w[tid]       + b[tid];
  out[row * D_MODEL + tid + 256] = (v1 - mean) * rstd * w[tid + 256] + b[tid + 256];
}

// ---------------------------------------------------------------------------
// fp32-emulated GEMM on bf16 WMMA:  C[m,n] = sum_k A[m*lda+k] * W[n*K+k]
// split each operand f = hi + lo (bf16 each), acc += hi*hi + hi*lo + lo*hi.
// Each wave owns a 16x32 output strip (two 16x16 N-tiles) so the A hi/lo
// fragments are converted once and reused across 6 WMMAs per K-step.
// K multiple of 32; N multiple of 32; M multiple of 16.
// epilogue: 0 = store, 1 = softplus(acc + bias[n]) via fast v_exp/v_log
// ---------------------------------------------------------------------------
__device__ __forceinline__ float softplus_f(float x) {
  // fast hw-trans softplus; exact-enough for f32 (feeds an exp downstream)
  return (x > 20.f) ? x : __logf(1.f + __expf(x));
}

__device__ __forceinline__ void pack8_split(v16bf& hi, v16bf& lo, int base,
                                            float4 f0, float4 f1) {
  float t[8] = {f0.x, f0.y, f0.z, f0.w, f1.x, f1.y, f1.z, f1.w};
#pragma unroll
  for (int i = 0; i < 8; ++i) {
    __bf16 h = (__bf16)t[i];
    hi[base + i] = h;
    lo[base + i] = (__bf16)(t[i] - (float)h);
  }
}

__global__ void gemm_wmma_f32split(const float* __restrict__ A, int lda,
                                   const float* __restrict__ W,
                                   float* __restrict__ C,
                                   int M, int N, int K,
                                   const float* __restrict__ bias,
                                   int epilogue) {
  const int wave = (int)((blockIdx.x * blockDim.x + threadIdx.x) >> 5);
  const int lane = threadIdx.x & 31;
  const int tilesN2 = N >> 5;                   // pairs of 16-wide N tiles
  const int tilesM  = M >> 4;
  if (wave >= tilesM * tilesN2) return;         // wave-uniform: EXEC stays all-1s
  const int tm  = wave / tilesN2;
  const int tnp = wave % tilesN2;
  const int half = lane >> 4;                   // 0 or 1
  const int r16  = lane & 15;

  const float* Arow  = A + (size_t)(tm * 16 + r16) * lda;    // A row per lane
  const float* Wrow0 = W + (size_t)(tnp * 32 + r16) * K;     // out col n
  const float* Wrow1 = Wrow0 + (size_t)16 * K;               // out col n+16

  v8f acc0 = {}, acc1 = {};
  for (int k0 = 0; k0 < K; k0 += 32) {
    __builtin_prefetch(Wrow0 + k0 + 64, 0, 1);  // global_prefetch_b8 hints
    __builtin_prefetch(Wrow1 + k0 + 64, 0, 1);
    // A fragment: elem i -> K=k0+16*half+i (matches B)
    const float4* ap0 = (const float4*)(Arow + k0 + half * 16);
    const float4* ap1 = (const float4*)(Arow + k0 + half * 16 + 8);
    float4 a0 = ap0[0], a1 = ap0[1], a2 = ap1[0], a3 = ap1[1];
    // B fragments (same k order as A): elem i -> K=k0+16*half+i at column n
    const float4* bp0 = (const float4*)(Wrow0 + k0 + half * 16);
    const float4* bp1 = (const float4*)(Wrow1 + k0 + half * 16);
    float4 b00 = bp0[0], b01 = bp0[1], b02 = bp0[2], b03 = bp0[3];
    float4 b10 = bp1[0], b11 = bp1[1], b12 = bp1[2], b13 = bp1[3];

    v16bf ah, al, bh0, bl0, bh1, bl1;
    pack8_split(ah, al, 0, a0, a1);
    pack8_split(ah, al, 8, a2, a3);
    pack8_split(bh0, bl0, 0, b00, b01);
    pack8_split(bh0, bl0, 8, b02, b03);
    pack8_split(bh1, bl1, 0, b10, b11);
    pack8_split(bh1, bl1, 8, b12, b13);

    acc0 = __builtin_amdgcn_wmma_f32_16x16x32_bf16(false, ah, false, bh0, (short)0, acc0, false, false);
    acc1 = __builtin_amdgcn_wmma_f32_16x16x32_bf16(false, ah, false, bh1, (short)0, acc1, false, false);
    acc0 = __builtin_amdgcn_wmma_f32_16x16x32_bf16(false, ah, false, bl0, (short)0, acc0, false, false);
    acc1 = __builtin_amdgcn_wmma_f32_16x16x32_bf16(false, ah, false, bl1, (short)0, acc1, false, false);
    acc0 = __builtin_amdgcn_wmma_f32_16x16x32_bf16(false, al, false, bh0, (short)0, acc0, false, false);
    acc1 = __builtin_amdgcn_wmma_f32_16x16x32_bf16(false, al, false, bh1, (short)0, acc1, false, false);
  }

  const int outn0 = tnp * 32 + r16;             // D layout: n = lane&15
  const int outn1 = outn0 + 16;
#pragma unroll
  for (int r = 0; r < 8; ++r) {                 // m = r + 8*(lane>=16)
    const int outm = tm * 16 + half * 8 + r;
    float v0 = acc0[r];
    float v1 = acc1[r];
    if (epilogue == 1) {
      v0 = softplus_f(v0 + bias[outn0]);
      v1 = softplus_f(v1 + bias[outn1]);
    }
    C[(size_t)outm * N + outn0] = v0;
    C[(size_t)outm * N + outn1] = v1;
  }
}

// ---------------------------------------------------------------------------
// depthwise causal conv (K=4) + SiLU on the x half of xz
// ---------------------------------------------------------------------------
__global__ void conv_silu_kernel(const float* __restrict__ xz,
                                 const float* __restrict__ cw,
                                 const float* __restrict__ cb,
                                 float* __restrict__ xout) {
  const int idx = blockIdx.x * blockDim.x + threadIdx.x;
  if (idx >= ROWS * D_INNER) return;
  const int d   = idx & (D_INNER - 1);
  const int row = idx >> 10;
  const int b   = row / SEQLEN;
  const int l   = row - b * SEQLEN;
  float acc = cb[d];
#pragma unroll
  for (int j = 0; j < D_CONV; ++j) {
    const int li = l - (D_CONV - 1) + j;
    if (li >= 0)
      acc += cw[d * D_CONV + j] * xz[(size_t)(b * SEQLEN + li) * (2 * D_INNER) + d];
  }
  xout[idx] = acc / (1.f + __expf(-acc));       // silu
}

// ---------------------------------------------------------------------------
// selective scan: one thread per (b, d), 16 states in registers,
// fused +D*x and *silu(z) epilogue
// ---------------------------------------------------------------------------
__global__ void scan_kernel(const float* __restrict__ dt,
                            const float* __restrict__ xc,
                            const float* __restrict__ xdbl,
                            const float* __restrict__ xz,
                            const float* __restrict__ A_log,
                            const float* __restrict__ Dp,
                            float* __restrict__ y) {
  const int d = blockIdx.x * blockDim.x + threadIdx.x;   // 0..1023
  const int b = blockIdx.y;                              // 0..3
  float a[D_STATE], h[D_STATE];
#pragma unroll
  for (int n = 0; n < D_STATE; ++n) {
    a[n] = -__expf(A_log[d * D_STATE + n]);
    h[n] = 0.f;
  }
  const float dsk = Dp[d];
  for (int l = 0; l < SEQLEN; ++l) {
    const int row = b * SEQLEN + l;
    const float dtv = dt[(size_t)row * D_INNER + d];
    const float xv  = xc[(size_t)row * D_INNER + d];
    const float dBx = dtv * xv;
    const float* bc = xdbl + (size_t)row * (DT_RANK + 2 * D_STATE);
    float acc = 0.f;
#pragma unroll
    for (int n = 0; n < D_STATE; ++n) {
      h[n] = __expf(dtv * a[n]) * h[n] + dBx * bc[DT_RANK + n];
      acc += h[n] * bc[DT_RANK + D_STATE + n];
    }
    acc += dsk * xv;
    const float zv = xz[(size_t)row * (2 * D_INNER) + D_INNER + d];
    y[(size_t)row * D_INNER + d] = acc * (zv / (1.f + __expf(-zv)));
  }
}

// ---------------------------------------------------------------------------
extern "C" void kernel_launch(void* const* d_in, const int* in_sizes, int n_in,
                              void* d_out, int out_size, void* d_ws, size_t ws_size,
                              hipStream_t stream) {
  (void)in_sizes; (void)n_in; (void)out_size; (void)ws_size;
  const float* x       = (const float*)d_in[0];
  const float* in_w    = (const float*)d_in[1];
  const float* conv_w  = (const float*)d_in[2];
  const float* conv_b  = (const float*)d_in[3];
  const float* xp_w    = (const float*)d_in[4];
  const float* dt_w    = (const float*)d_in[5];
  const float* dt_b    = (const float*)d_in[6];
  const float* A_log   = (const float*)d_in[7];
  const float* Dskip   = (const float*)d_in[8];
  const float* out_w   = (const float*)d_in[9];
  const float* norm_w  = (const float*)d_in[10];
  const float* norm_b  = (const float*)d_in[11];
  const float* normf_w = (const float*)d_in[12];
  const float* normf_b = (const float*)d_in[13];

  float* ws = (float*)d_ws;
  float* residual = ws; ws += ROWS * D_MODEL;
  float* hn       = ws; ws += ROWS * D_MODEL;
  float* hidden   = ws; ws += ROWS * D_MODEL;
  float* xzb      = ws; ws += ROWS * 2 * D_INNER;
  float* xcb      = ws; ws += ROWS * D_INNER;
  float* xdbl     = ws; ws += ROWS * (DT_RANK + 2 * D_STATE);
  float* dtb      = ws; ws += ROWS * D_INNER;
  float* yb       = ws; ws += ROWS * D_INNER;

  for (int i = 0; i < DEPTH; ++i) {
    add_ln_kernel<<<ROWS, 256, 0, stream>>>(
        x, hidden, residual, hn,
        norm_w + (size_t)i * D_MODEL, norm_b + (size_t)i * D_MODEL, i == 0);

    {  // xz = hn @ in_proj_w^T   [784,512]x[512,2048]
      int waves = (ROWS / 16) * (2 * D_INNER / 32);
      gemm_wmma_f32split<<<(waves + 7) / 8, 256, 0, stream>>>(
          hn, D_MODEL, in_w + (size_t)i * 2 * D_INNER * D_MODEL,
          xzb, ROWS, 2 * D_INNER, D_MODEL, nullptr, 0);
    }

    conv_silu_kernel<<<(ROWS * D_INNER + 255) / 256, 256, 0, stream>>>(
        xzb, conv_w + (size_t)i * D_INNER * D_CONV,
        conv_b + (size_t)i * D_INNER, xcb);

    {  // x_dbl = x @ x_proj_w^T   [784,1024]x[1024,64]
      int waves = (ROWS / 16) * ((DT_RANK + 2 * D_STATE) / 32);
      gemm_wmma_f32split<<<(waves + 7) / 8, 256, 0, stream>>>(
          xcb, D_INNER, xp_w + (size_t)i * (DT_RANK + 2 * D_STATE) * D_INNER,
          xdbl, ROWS, DT_RANK + 2 * D_STATE, D_INNER, nullptr, 0);
    }

    {  // dt = softplus(dt_in @ dt_proj_w^T + dt_bias)  [784,32]x[32,1024]
      int waves = (ROWS / 16) * (D_INNER / 32);
      gemm_wmma_f32split<<<(waves + 7) / 8, 256, 0, stream>>>(
          xdbl, DT_RANK + 2 * D_STATE, dt_w + (size_t)i * D_INNER * DT_RANK,
          dtb, ROWS, D_INNER, DT_RANK, dt_b + (size_t)i * D_INNER, 1);
    }

    scan_kernel<<<dim3(D_INNER / 256, BATCH), 256, 0, stream>>>(
        dtb, xcb, xdbl, xzb,
        A_log + (size_t)i * D_INNER * D_STATE,
        Dskip + (size_t)i * D_INNER, yb);

    {  // hidden = y @ out_proj_w^T  [784,1024]x[1024,512]
      int waves = (ROWS / 16) * (D_MODEL / 32);
      gemm_wmma_f32split<<<(waves + 7) / 8, 256, 0, stream>>>(
          yb, D_INNER, out_w + (size_t)i * D_MODEL * D_INNER,
          hidden, ROWS, D_MODEL, D_INNER, nullptr, 0);
    }
  }

  // final: residual += hidden; out = LN(residual, normf)
  add_ln_kernel<<<ROWS, 256, 0, stream>>>(
      x, hidden, residual, (float*)d_out, normf_w, normf_b, 0);
}
